// KANLayer_61065845015250
// MI455X (gfx1250) — compile-verified, matrix operand order checked by reading, not run on hardware
//
#include <hip/hip_runtime.h>

typedef __attribute__((ext_vector_type(16))) _Float16 v16h;
typedef __attribute__((ext_vector_type(8)))  _Float16 v8h;
typedef __attribute__((ext_vector_type(8)))  float    v8f;

#define IN_DIM   256
#define OUT_DIM  256
#define BATCH    32768
#define NB       9
#define KSLOT    16                 // padded K slots per input dim
#define KPAD     (IN_DIM * KSLOT)   // 4096
#define M_TILE   128
#define N_TILE   128
#define LDS_STRIDE 40               // 32 halves of data + 8 pad (bank spread)

// Degree-3 open-uniform B-spline basis on knots
// [-1 x4, -2/3, -1/3, 0, 1/3, 2/3, 1 x4]; 9 basis functions.
// Knots are compile-time constants: all division guards and reciprocals fold.
__device__ __forceinline__ void bspline9(float x, float* __restrict__ Bv) {
    constexpr float T[13] = {
        -1.0f, -1.0f, -1.0f, -1.0f,
        -2.0f / 3.0f, -1.0f / 3.0f, 0.0f, 1.0f / 3.0f, 2.0f / 3.0f,
         1.0f,  1.0f,  1.0f,  1.0f
    };
    float b[12];
#pragma unroll
    for (int s = 0; s < 12; ++s) {
        float v = (x >= T[s] && x < T[s + 1]) ? 1.0f : 0.0f;
        // include right endpoint x == x_max in last non-degenerate span
        if (T[s] < T[s + 1] && T[s + 1] >= T[12]) {
            if (x >= T[12]) v = 1.0f;
        }
        b[s] = v;
    }
#pragma unroll
    for (int d = 1; d <= 3; ++d) {
#pragma unroll
        for (int s = 0; s < 12 - d; ++s) {
            float d1 = T[s + d] - T[s];
            float d2 = T[s + d + 1] - T[s + 1];
            float w1 = (d1 > 0.0f) ? (x - T[s]) / d1 : 0.0f;
            float w2 = (d2 > 0.0f) ? (T[s + d + 1] - x) / d2 : 0.0f;
            b[s] = w1 * b[s] + w2 * b[s + 1];
        }
    }
#pragma unroll
    for (int r = 0; r < NB; ++r) Bv[r] = b[r];
}

// Repack coeff (OUT, IN, 9) f32 -> coeffT (OUT=N, KPAD) f16, K-contiguous,
// zero-padded so each input dim owns 16 K slots (slots 9..15 are zero).
__global__ void kan_prep_coeff(const float* __restrict__ coeff,
                               _Float16* __restrict__ coeffT) {
    int idx = blockIdx.x * blockDim.x + threadIdx.x;  // 0 .. OUT_DIM*KPAD-1
    int n  = idx >> 12;                               // / KPAD
    int kp = idx & (KPAD - 1);
    int i  = kp >> 4;
    int r  = kp & 15;
    float v = (r < NB) ? coeff[(n * IN_DIM + i) * NB + r] : 0.0f;
    coeffT[(size_t)n * KPAD + kp] = (_Float16)v;
}

// Fused: build basis A-tile in LDS per K-step, WMMA against coeffT, add bias.
__global__ void __launch_bounds__(256)
kan_wmma(const float* __restrict__ x,
         const _Float16* __restrict__ coeffT,
         const float* __restrict__ bias,
         float* __restrict__ out) {
    __shared__ _Float16 Atile[M_TILE * LDS_STRIDE];

    const int tid  = threadIdx.x;
    const int lane = tid & 31;
    const int wave = tid >> 5;
    const int mrow = lane & 15;   // row within 16x16 sub-tile / n within B frag
    const int half = lane >> 4;   // K-half selector per ISA 16-bit A/B layout

    const int m0 = blockIdx.y * M_TILE;
    const int n0 = blockIdx.x * N_TILE;

    v8f acc[8];
#pragma unroll
    for (int s = 0; s < 8; ++s) acc[s] = (v8f){};

    const int wrow = tid >> 1;  // 0..127: row of A tile this thread produces
    const int wii  = tid & 1;   // which of the 2 input dims in this K-step

    for (int ks = 0; ks < KPAD / 32; ++ks) {
        // ---- produce A tile (128 rows x 32 K-halves) into LDS ----
        {
            const int   i  = ks * 2 + wii;
            const float xv = x[(size_t)(m0 + wrow) * IN_DIM + i];
            float Bv[NB];
            bspline9(xv, Bv);
            v8h lo, hi8 = {};
#pragma unroll
            for (int r = 0; r < 8; ++r) lo[r] = (_Float16)Bv[r];
            hi8[0] = (_Float16)Bv[8];           // slots 9..15 stay zero (pad)
            _Float16* dst = &Atile[wrow * LDS_STRIDE + wii * 16];
            *(v8h*)(dst)     = lo;
            *(v8h*)(dst + 8) = hi8;
        }
        __syncthreads();

        // ---- A fragment: lanes 0-15 K=0-7 & 16-23, lanes 16-31 K=8-15 & 24-31
        const _Float16* arow = &Atile[(wave * 16 + mrow) * LDS_STRIDE];
        v8h alo = *(const v8h*)(arow + half * 8);
        v8h ahi = *(const v8h*)(arow + 16 + half * 8);
        v16h afrag = __builtin_shufflevector(alo, ahi,
                        0, 1, 2, 3, 4, 5, 6, 7, 8, 9, 10, 11, 12, 13, 14, 15);

        // ---- 8 B fragments from coeffT + 8 WMMAs (16x16 n-subtiles) ----
        const int kbase = ks * 32;
#pragma unroll
        for (int s = 0; s < 8; ++s) {
            const int n = n0 + s * 16 + mrow;
            const _Float16* brow = coeffT + (size_t)n * KPAD + kbase;
            v8h blo = *(const v8h*)(brow + half * 8);
            v8h bhi = *(const v8h*)(brow + 16 + half * 8);
            v16h bfrag = __builtin_shufflevector(blo, bhi,
                            0, 1, 2, 3, 4, 5, 6, 7, 8, 9, 10, 11, 12, 13, 14, 15);
            acc[s] = __builtin_amdgcn_wmma_f32_16x16x32_f16(
                         false, afrag, false, bfrag,
                         (short)0, acc[s], false, false);
        }
        __syncthreads();  // protect LDS A-tile before next K-step rewrites it
    }

    // ---- epilogue: C/D layout m = half*8 + vgpr, n = mrow; add bias ----
#pragma unroll
    for (int s = 0; s < 8; ++s) {
        const int   col = n0 + s * 16 + mrow;
        const float bv  = bias[col];
#pragma unroll
        for (int r = 0; r < 8; ++r) {
            const int row = m0 + wave * 16 + half * 8 + r;
            out[(size_t)row * OUT_DIM + col] = acc[s][r] + bv;
        }
    }
}

extern "C" void kernel_launch(void* const* d_in, const int* in_sizes, int n_in,
                              void* d_out, int out_size, void* d_ws, size_t ws_size,
                              hipStream_t stream) {
    const float* x     = (const float*)d_in[0];   // (32768, 256) f32
    const float* coeff = (const float*)d_in[1];   // (256, 256, 9) f32
    const float* bias  = (const float*)d_in[2];   // (256,) f32
    float*       out   = (float*)d_out;           // (32768, 256) f32
    _Float16*    coeffT = (_Float16*)d_ws;        // (256, 4096) f16 = 2 MB

    // 1) repack + pad coefficients (runs every call; deterministic)
    kan_prep_coeff<<<dim3((OUT_DIM * KPAD) / 256), dim3(256), 0, stream>>>(coeff, coeffT);

    // 2) fused basis + WMMA GEMM
    dim3 grid(OUT_DIM / N_TILE, BATCH / M_TILE);  // (2, 256)
    kan_wmma<<<grid, dim3(256), 0, stream>>>(x, coeffT, bias, out);
}
